// Relative_Transformer_65609920414311
// MI455X (gfx1250) — matrix operand order repaired and checked by reading, not compiled, not run on hardware
//
#include <hip/hip_runtime.h>
#include <hip/hip_bf16.h>
#include <math.h>

typedef __attribute__((ext_vector_type(16))) __bf16 v16bf;
typedef __attribute__((ext_vector_type(8)))  __bf16 v8bf;
typedef __attribute__((ext_vector_type(2)))  __bf16 v2bf;
typedef __attribute__((ext_vector_type(8)))  float  v8f;
typedef __attribute__((ext_vector_type(4)))  unsigned int v4u;
typedef __attribute__((ext_vector_type(8)))  int v8i;
typedef __attribute__((ext_vector_type(4)))  int v4i;

#define TM 128
#define TN 64
#define KS 32
#define LDSS 40   // LDS row stride (bf16): 64B data + 16B pad -> matches TDM pad cfg

#if __has_include(<hip/amd_detail/amd_gfx1250_TDM.h>)
#define TDM_6ARG 1
#else
#define TDM_6ARG 0
#endif

// Issue a TDM 2D tile load: rows x cols (bf16 elems), row stride in elems,
// into LDS at lds_off, with 16B padding after each 64B row chunk.
static __device__ inline void tdm_issue_2d(const void* gptr, unsigned lds_off,
                                           int rows, int cols, long long stride_elems)
{
  unsigned long long ga = (unsigned long long)gptr;
  v4u g0;
  g0[0] = 1u;                                            // count=1, user mode
  g0[1] = lds_off;                                       // LDS byte address
  g0[2] = (unsigned)ga;                                  // global_addr[31:0]
  g0[3] = (unsigned)((ga >> 32) & 0x01FFFFFFu) | (2u << 30); // addr[56:32], type=2
  v8i g1;
  // data_size=1 (2B), pad_enable=1, pad_interval=3 (16 DWORDs), pad_amount=3 (4 DWORDs)
  g1[0] = (1 << 16) | (1 << 20) | (3 << 22) | (3 << 25);
  const unsigned td0 = 0x40000000u, td1 = 0x40000000u;   // huge tensor dims: no OOB clip
  g1[1] = (int)((td0 & 0xFFFFu) << 16);
  g1[2] = (int)((td0 >> 16) | ((td1 & 0xFFFFu) << 16));
  g1[3] = (int)((td1 >> 16) | ((unsigned)cols << 16));   // tile_dim0 = cols
  g1[4] = rows;                                          // tile_dim1 = rows, tile_dim2=0
  unsigned long long s0 = (unsigned long long)stride_elems;
  g1[5] = (int)(s0 & 0xFFFFFFFFu);
  g1[6] = (int)((s0 >> 32) & 0xFFFFu);                   // dim1_stride = 0 (2D)
  g1[7] = 0;
  v4i z4 = {};
#if TDM_6ARG
  v8i z8 = {};
  __builtin_amdgcn_tensor_load_to_lds(g0, g1, z4, z4, z8, 0);
#else
  __builtin_amdgcn_tensor_load_to_lds(g0, g1, z4, z4, 0);
#endif
}

static __device__ inline unsigned lds_off32(const void* p)
{
  return (unsigned)(unsigned long long)p;   // low 32 bits of shared pointer = LDS offset
}

// Generic tiled WMMA GEMM on bf16 inputs (128x64 block tile, 32x32 wave tiles):
//   out = act( sum_k A[m,k]*B'[k,n] + bias[n] + resid[m,n] ), out -> fp32 and/or bf16
// transB==0: B' = B^T, B row-major [N,K]  (A and B tiles DMA'd by TDM)
// transB==1: B' = B,   B row-major [K,N]  (B staged manually; A by TDM)
// Requires M%128==0, N%64==0, K%32==0.
__global__ __launch_bounds__(256)
void gemm_wmma_kernel(const __bf16* __restrict__ A, int lda, long long sAo, long long sAi,
                      const __bf16* __restrict__ Bm, int ldb, long long sBo, long long sBi,
                      int transB,
                      float* __restrict__ Cfp, __bf16* __restrict__ Cbf,
                      int ldc, long long sCo, long long sCi,
                      const float* __restrict__ bias, const float* __restrict__ resid,
                      int M, int N, int K, int innerB, int act)
{
  const int z  = blockIdx.z;
  const int zo = z / innerB;
  const int zi = z - zo * innerB;
  A  += zo * sAo + zi * sAi;
  Bm += zo * sBo + zi * sBi;
  if (Cfp)   Cfp   += zo * sCo + zi * sCi;
  if (Cbf)   Cbf   += zo * sCo + zi * sCi;
  if (resid) resid += zo * sCo + zi * sCi;

  __shared__ __align__(16) __bf16 As[2][TM][LDSS];
  __shared__ __align__(16) __bf16 Bs[2][TN][LDSS];

  const int tid  = threadIdx.x;
  const int lane = tid & 31;
  const int wave = tid >> 5;
  const int wm   = (wave >> 1) * 32;   // 4 row groups of 32
  const int wn   = (wave & 1) * 32;    // 2 col groups of 32
  const int blockM = blockIdx.y * TM;
  const int blockN = blockIdx.x * TN;
  const int mrow = lane & 15;
  const int hf   = lane >> 4;

  v8f acc00 = {}, acc01 = {}, acc10 = {}, acc11 = {};

  __bf16 bT[8];   // manual staging regs for transB B tiles

  auto loadB_nn = [&](int k0) {
    #pragma unroll
    for (int i = 0; i < 4; ++i) {
      int flat = tid + 256 * i;
      int n = flat & 63, p2 = flat >> 6;
      const __bf16* bp = &Bm[(long long)(k0 + 2 * p2) * ldb + (blockN + n)];
      bT[2 * i]     = bp[0];
      bT[2 * i + 1] = bp[ldb];
    }
  };
  auto storeB_nn = [&](int buf) {
    #pragma unroll
    for (int i = 0; i < 4; ++i) {
      int flat = tid + 256 * i;
      int n = flat & 63, p2 = flat >> 6;
      v2bf pk;
      pk[0] = bT[2 * i];
      pk[1] = bT[2 * i + 1];
      *(v2bf*)&Bs[buf][n][2 * p2] = pk;
    }
  };

  // prologue: DMA tile 0
  if (tid < 32) {
    tdm_issue_2d(&A[(long long)blockM * lda], lds_off32(&As[0][0][0]), TM, KS, lda);
    if (!transB)
      tdm_issue_2d(&Bm[(long long)blockN * ldb], lds_off32(&Bs[0][0][0]), TN, KS, ldb);
  }
  if (transB) loadB_nn(0);

  int buf = 0;
  for (int k0 = 0; k0 < K; k0 += KS, buf ^= 1) {
    const bool more = (k0 + KS) < K;
    if (tid < 32) {
      if (more) {
        tdm_issue_2d(&A[(long long)blockM * lda + (k0 + KS)],
                     lds_off32(&As[buf ^ 1][0][0]), TM, KS, lda);
        if (!transB)
          tdm_issue_2d(&Bm[(long long)blockN * ldb + (k0 + KS)],
                       lds_off32(&Bs[buf ^ 1][0][0]), TN, KS, ldb);
      }
      if (more) {
        if (transB) __builtin_amdgcn_s_wait_tensorcnt(1);
        else        __builtin_amdgcn_s_wait_tensorcnt(2);
      } else {
        __builtin_amdgcn_s_wait_tensorcnt(0);
      }
    }
    if (transB) {
      storeB_nn(buf);
      if (more) loadB_nn(k0 + KS);
    }
    __syncthreads();

    // A fragments = two contiguous aligned 8-elem runs: K in [8*hf,+8) and [16+8*hf,+8)
    const v8bf* ap0 = (const v8bf*)&As[buf][wm + mrow][0];       // 5 v8bf per row
    const v8bf* ap1 = (const v8bf*)&As[buf][wm + 16 + mrow][0];
    v16bf afrag0 = __builtin_shufflevector(ap0[hf], ap0[2 + hf],
        0, 1, 2, 3, 4, 5, 6, 7, 8, 9, 10, 11, 12, 13, 14, 15);
    v16bf afrag1 = __builtin_shufflevector(ap1[hf], ap1[2 + hf],
        0, 1, 2, 3, 4, 5, 6, 7, 8, 9, 10, 11, 12, 13, 14, 15);

    // B fragments = two contiguous aligned 8-elem runs: K in [8*hf,+8) and [16+8*hf,+8)
    const v8bf* bp0 = (const v8bf*)&Bs[buf][wn + mrow][0];
    const v8bf* bp1 = (const v8bf*)&Bs[buf][wn + 16 + mrow][0];
    v16bf bfrag0 = __builtin_shufflevector(bp0[hf], bp0[2 + hf],
        0, 1, 2, 3, 4, 5, 6, 7, 8, 9, 10, 11, 12, 13, 14, 15);
    v16bf bfrag1 = __builtin_shufflevector(bp1[hf], bp1[2 + hf],
        0, 1, 2, 3, 4, 5, 6, 7, 8, 9, 10, 11, 12, 13, 14, 15);

    acc00 = __builtin_amdgcn_wmma_f32_16x16x32_bf16(false, afrag0, false, bfrag0,
                                                    (short)0, acc00, false, false);
    acc01 = __builtin_amdgcn_wmma_f32_16x16x32_bf16(false, afrag0, false, bfrag1,
                                                    (short)0, acc01, false, false);
    acc10 = __builtin_amdgcn_wmma_f32_16x16x32_bf16(false, afrag1, false, bfrag0,
                                                    (short)0, acc10, false, false);
    acc11 = __builtin_amdgcn_wmma_f32_16x16x32_bf16(false, afrag1, false, bfrag1,
                                                    (short)0, acc11, false, false);
    __syncthreads();
  }

  // C/D layout: lane N = lane&15; VGPR v -> M = v + 8*(lane>>4)
  #pragma unroll
  for (int vv = 0; vv < 8; ++vv) {
    int m0 = blockM + wm + vv + 8 * hf;
    int m1 = m0 + 16;
    int n0 = blockN + wn + mrow;
    int n1 = n0 + 16;
    float v00 = acc00[vv], v01 = acc01[vv], v10 = acc10[vv], v11 = acc11[vv];
    if (bias)  { v00 += bias[n0]; v01 += bias[n1]; v10 += bias[n0]; v11 += bias[n1]; }
    if (resid) {
      v00 += resid[(long long)m0 * ldc + n0];
      v01 += resid[(long long)m0 * ldc + n1];
      v10 += resid[(long long)m1 * ldc + n0];
      v11 += resid[(long long)m1 * ldc + n1];
    }
    if (act) {
      v00 = fmaxf(v00, 0.f); v01 = fmaxf(v01, 0.f);
      v10 = fmaxf(v10, 0.f); v11 = fmaxf(v11, 0.f);
    }
    if (Cfp) {
      Cfp[(long long)m0 * ldc + n0] = v00;
      Cfp[(long long)m0 * ldc + n1] = v01;
      Cfp[(long long)m1 * ldc + n0] = v10;
      Cfp[(long long)m1 * ldc + n1] = v11;
    }
    if (Cbf) {
      Cbf[(long long)m0 * ldc + n0] = (__bf16)v00;
      Cbf[(long long)m0 * ldc + n1] = (__bf16)v01;
      Cbf[(long long)m1 * ldc + n0] = (__bf16)v10;
      Cbf[(long long)m1 * ldc + n1] = (__bf16)v11;
    }
  }
}

__global__ __launch_bounds__(256)
void cvt_bf16_kernel(const float* __restrict__ in, __bf16* __restrict__ out, long long n8)
{
  long long i = (long long)blockIdx.x * 256 + threadIdx.x;
  if (i >= n8) return;
  const float4* p = (const float4*)(in + i * 8);
  float4 a = p[0], b = p[1];
  v8bf pk;
  pk[0] = (__bf16)a.x; pk[1] = (__bf16)a.y; pk[2] = (__bf16)a.z; pk[3] = (__bf16)a.w;
  pk[4] = (__bf16)b.x; pk[5] = (__bf16)b.y; pk[6] = (__bf16)b.z; pk[7] = (__bf16)b.w;
  *(v8bf*)(out + i * 8) = pk;
}

__global__ __launch_bounds__(256)
void embed_kernel(const int* __restrict__ idx, const float* __restrict__ wte,
                  float* __restrict__ out, int C)
{
  long long row = blockIdx.x;
  const float* src = wte + (long long)idx[row] * C;
  float* dst = out + row * C;
  for (int i = threadIdx.x; i < C; i += blockDim.x) dst[i] = src[i];
}

// LayerNorm: fp32 in -> bf16 out
__global__ __launch_bounds__(256)
void ln_kernel(const float* __restrict__ x, const float* __restrict__ w,
               const float* __restrict__ b, __bf16* __restrict__ out, int C)
{
  long long row = blockIdx.x;
  const float* xr = x + row * C;
  __bf16* orow = out + row * C;
  int tid = threadIdx.x;
  float s = 0.f, ss = 0.f;
  for (int i = tid; i < C; i += 256) { float v = xr[i]; s += v; ss += v * v; }
  __shared__ float r1[256], r2[256];
  r1[tid] = s; r2[tid] = ss;
  __syncthreads();
  for (int st = 128; st > 0; st >>= 1) {
    if (tid < st) { r1[tid] += r1[tid + st]; r2[tid] += r2[tid + st]; }
    __syncthreads();
  }
  float mean = r1[0] / C;
  float var  = r2[0] / C - mean * mean;
  float inv  = rsqrtf(var + 1e-5f);
  for (int i = tid; i < C; i += 256)
    orow[i] = (__bf16)((xr[i] - mean) * inv * w[i] + b[i]);
}

// probs[q,k] = softmax_k( (S[q,k] + P[q, q-k]) * scale , causal ), bf16 out
__global__ __launch_bounds__(256)
void softmax_rel_kernel(const float* __restrict__ S, const float* __restrict__ P,
                        __bf16* __restrict__ probs, int T, float scale)
{
  int q = blockIdx.x;
  long long base = (long long)blockIdx.y * T * T + (long long)q * T;
  const float* srow = S + base;
  const float* prow = P + base;
  __bf16* orow = probs + base;
  int tid = threadIdx.x;
  __shared__ float buf[512];
  __shared__ float red[256];
  for (int k = tid; k < T; k += 256)
    buf[k] = (k <= q) ? (srow[k] + prow[q - k]) * scale : -INFINITY;
  __syncthreads();
  float m = -INFINITY;
  for (int k = tid; k < T; k += 256) m = fmaxf(m, buf[k]);
  red[tid] = m;
  __syncthreads();
  for (int st = 128; st > 0; st >>= 1) {
    if (tid < st) red[tid] = fmaxf(red[tid], red[tid + st]);
    __syncthreads();
  }
  m = red[0];
  __syncthreads();
  float sum = 0.f;
  for (int k = tid; k < T; k += 256) {
    float e = __expf(buf[k] - m);
    buf[k] = e;
    sum += e;
  }
  red[tid] = sum;
  __syncthreads();
  for (int st = 128; st > 0; st >>= 1) {
    if (tid < st) red[tid] += red[tid + st];
    __syncthreads();
  }
  float inv = 1.f / red[0];
  for (int k = tid; k < T; k += 256) orow[k] = (__bf16)(buf[k] * inv);
}

extern "C" void kernel_launch(void* const* d_in, const int* in_sizes, int n_in,
                              void* d_out, int out_size, void* d_ws, size_t ws_size,
                              hipStream_t stream)
{
  (void)in_sizes; (void)n_in; (void)out_size; (void)ws_size;
  const int Bq = 8, T = 512, Cc = 512, Vv = 32000;
  const int Mrows = Bq * T;                 // 4096
  const int heads[3] = {8, 1, 1};

  const int*   idx = (const int*)d_in[0];
  const float* wte = (const float*)d_in[1];
  auto LP = [&](int l, int j) -> const float* {
    return (const float*)d_in[2 + l * 13 + j];
  };
  const float* lnf_w   = (const float*)d_in[41];
  const float* lnf_b   = (const float*)d_in[42];
  const float* lm_head = (const float*)d_in[43];
  float* out = (float*)d_out;

  // workspace carving (256B aligned)
  char* cur = (char*)d_ws;
  auto alloc = [&](size_t bytes) -> void* {
    void* p = cur;
    cur += (bytes + 255) & ~(size_t)255;
    return p;
  };
  const long long MC = (long long)Mrows * Cc;
  const long long TT = (long long)T * T;
  float*  x     = (float*) alloc(MC * 4);
  __bf16* lnb   = (__bf16*)alloc(MC * 2);
  __bf16* qb    = (__bf16*)alloc(MC * 2);
  __bf16* kb    = (__bf16*)alloc(MC * 2);
  __bf16* vb    = (__bf16*)alloc(MC * 2);
  __bf16* yb    = (__bf16*)alloc(MC * 2);
  __bf16* h1    = (__bf16*)alloc(MC * 4 * 2);
  float*  S1    = (float*) alloc((long long)Bq * 8 * TT * 4);
  float*  Pb    = (float*) alloc((long long)Bq * 8 * TT * 4);
  __bf16* prob  = (__bf16*)alloc((long long)Bq * 8 * TT * 2);
  __bf16* wQ    = (__bf16*)alloc((long long)Cc * Cc * 2);
  __bf16* wK    = (__bf16*)alloc((long long)Cc * Cc * 2);
  __bf16* wV    = (__bf16*)alloc((long long)Cc * Cc * 2);
  __bf16* wP    = (__bf16*)alloc((long long)Cc * Cc * 2);
  __bf16* wpeb  = (__bf16*)alloc((long long)513 * Cc * 2);
  __bf16* wfc   = (__bf16*)alloc((long long)4 * Cc * Cc * 2);
  __bf16* wpj   = (__bf16*)alloc((long long)4 * Cc * Cc * 2);
  __bf16* wlm   = (__bf16*)alloc((long long)Vv * Cc * 2);

  auto cvt = [&](const float* src, __bf16* dst, long long n) {
    long long n8 = n / 8;
    cvt_bf16_kernel<<<(unsigned)((n8 + 255) / 256), 256, 0, stream>>>(src, dst, n8);
  };
  auto gemm = [&](const __bf16* A, int lda, long long sAo, long long sAi,
                  const __bf16* Bm, int ldb, long long sBo, long long sBi, int transB,
                  float* Cfp, __bf16* Cbf, int ldc, long long sCo, long long sCi,
                  const float* bias, const float* resid,
                  int Mi, int Ni, int Ki, int outer, int inner, int act) {
    dim3 grid(Ni / TN, Mi / TM, outer * inner);
    gemm_wmma_kernel<<<grid, 256, 0, stream>>>(A, lda, sAo, sAi,
                                               Bm, ldb, sBo, sBi, transB,
                                               Cfp, Cbf, ldc, sCo, sCi,
                                               bias, resid, Mi, Ni, Ki, inner, act);
  };

  // embedding (fp32 residual stream) + lm_head weight conversion
  embed_kernel<<<Mrows, 256, 0, stream>>>(idx, wte, x, Cc);
  cvt(lm_head, wlm, (long long)Vv * Cc);

  for (int l = 0; l < 3; ++l) {
    const int nh = heads[l];
    const int e  = Cc / nh;
    const float scale = 1.0f / sqrtf((float)e);

    // convert this layer's weights to bf16
    cvt(LP(l, 2), wQ, (long long)Cc * Cc);
    cvt(LP(l, 3), wK, (long long)Cc * Cc);
    cvt(LP(l, 4), wV, (long long)Cc * Cc);
    cvt(LP(l, 5), wP, (long long)Cc * Cc);
    cvt(LP(l, 6), wpeb, (long long)513 * Cc);
    cvt(LP(l, 9), wfc, (long long)4 * Cc * Cc);
    cvt(LP(l, 11), wpj, (long long)4 * Cc * Cc);

    // ln1 -> bf16
    ln_kernel<<<Mrows, 256, 0, stream>>>(x, LP(l, 0), LP(l, 1), lnb, Cc);
    // q,k,v (bf16 out)
    gemm(lnb, Cc, 0, 0, wQ, Cc, 0, 0, 0, nullptr, qb, Cc, 0, 0,
         nullptr, nullptr, Mrows, Cc, Cc, 1, 1, 0);
    gemm(lnb, Cc, 0, 0, wK, Cc, 0, 0, 0, nullptr, kb, Cc, 0, 0,
         nullptr, nullptr, Mrows, Cc, Cc, 1, 1, 0);
    gemm(lnb, Cc, 0, 0, wV, Cc, 0, 0, 0, nullptr, vb, Cc, 0, 0,
         nullptr, nullptr, Mrows, Cc, Cc, 1, 1, 0);

    // S1[b,h] = q @ k^T ; P[b,h][q,d] = q @ wpe_h^T   (fp32 out)
    gemm(qb, Cc, (long long)T * Cc, e, kb, Cc, (long long)T * Cc, e, 0,
         S1, nullptr, T, (long long)nh * TT, TT,
         nullptr, nullptr, T, T, e, Bq, nh, 0);
    gemm(qb, Cc, (long long)T * Cc, e, wpeb, Cc, 0, e, 0,
         Pb, nullptr, T, (long long)nh * TT, TT,
         nullptr, nullptr, T, T, e, Bq, nh, 0);

    softmax_rel_kernel<<<dim3(T, Bq * nh), 256, 0, stream>>>(S1, Pb, prob, T, scale);

    // y[b,h] = probs @ v  (NN, bf16 out)
    gemm(prob, T, (long long)nh * TT, TT, vb, Cc, (long long)T * Cc, e, 1,
         nullptr, yb, Cc, (long long)T * Cc, e,
         nullptr, nullptr, T, e, T, Bq, nh, 0);

    // x = y @ Pw^T + x  (fp32)
    gemm(yb, Cc, 0, 0, wP, Cc, 0, 0, 0, x, nullptr, Cc, 0, 0,
         nullptr, x, Mrows, Cc, Cc, 1, 1, 0);

    // MLP
    ln_kernel<<<Mrows, 256, 0, stream>>>(x, LP(l, 7), LP(l, 8), lnb, Cc);
    gemm(lnb, Cc, 0, 0, wfc, Cc, 0, 0, 0, nullptr, h1, 4 * Cc, 0, 0,
         LP(l, 10), nullptr, Mrows, 4 * Cc, Cc, 1, 1, 1);
    gemm(h1, 4 * Cc, 0, 0, wpj, 4 * Cc, 0, 0, 0, x, nullptr, Cc, 0, 0,
         LP(l, 12), x, Mrows, Cc, 4 * Cc, 1, 1, 0);
  }

  // final LN + lm_head
  ln_kernel<<<Mrows, 256, 0, stream>>>(x, lnf_w, lnf_b, lnb, Cc);
  gemm(lnb, Cc, 0, 0, wlm, Cc, 0, 0, 0, out, nullptr, Vv, 0, 0,
       nullptr, nullptr, Mrows, Vv, Cc, 1, 1, 0);
}
